// SimplestGNN_23252952940861
// MI455X (gfx1250) — matrix operand order repaired and checked by reading, not compiled, not run on hardware
//
#include <hip/hip_runtime.h>
#include <hip/hip_bf16.h>
#include <stdint.h>

// ---------------------------------------------------------------------------
// GCN forward for MI455X (gfx1250, wave32).
// GEMMs run on v_wmma_f32_16x16x32_bf16 (fp32 inputs converted once to bf16,
// f32 accumulate). Edge aggregation = L2-resident f32 atomics (51MB features
// fit in the 192MB L2). Weights staged in LDS (<=57KB of 320KB/WGP).
// GEMM wave tiling is 2x4 (M x N 16x16 tiles): each LDS B-fragment feeds two
// WMMAs, halving ds_load pressure vs a 1x8 tiling (LDS bw is the likely
// throttle with 8 waves/WGP issuing WMMA back-to-back).
// ---------------------------------------------------------------------------

typedef __attribute__((ext_vector_type(16))) __bf16 v16bf;
typedef __attribute__((ext_vector_type(8)))  float  v8f;

union FragAB { v16bf v; uint4 q[2]; };

__device__ __forceinline__ unsigned short f32_to_bf16_rne(float f) {
  union { float f; unsigned u; } v; v.f = f;
  unsigned r = v.u + 0x7FFFu + ((v.u >> 16) & 1u);
  return (unsigned short)(r >> 16);
}

// ---------------- degree ----------------
__global__ void k_deg_init(float* deg, int n) {
  int i = blockIdx.x * blockDim.x + threadIdx.x;
  if (i < n) deg[i] = 1.0f;  // self-loop
}
__global__ void k_deg_count(const int* __restrict__ dst, float* deg, int e) {
  int i = blockIdx.x * blockDim.x + threadIdx.x;
  if (i < e) atomicAdd(&deg[dst[i]], 1.0f);
}
__global__ void k_deg_fin(float* dsq, float* dinv, int n) {
  int i = blockIdx.x * blockDim.x + threadIdx.x;
  if (i < n) { float d = dsq[i]; dsq[i] = rsqrtf(d); dinv[i] = 1.0f / d; }
}

// ---------------- bf16 conversions ----------------
// x: [N, K] f32 row-major -> A: [N, Kpad] bf16 (zero padded)
__global__ void k_conv_a(const float* __restrict__ x, unsigned short* __restrict__ a,
                         int n, int k, int kpad) {
  long long i = (long long)blockIdx.x * blockDim.x + threadIdx.x;
  long long tot = (long long)n * kpad;
  if (i >= tot) return;
  int row = (int)(i / kpad), c = (int)(i % kpad);
  a[i] = (c < k) ? f32_to_bf16_rne(x[(long long)row * k + c]) : (unsigned short)0;
}
// W: [K, 128] f32 row-major -> Wt: [128, Kpad] bf16 (transposed, zero padded)
__global__ void k_conv_wt(const float* __restrict__ w, unsigned short* __restrict__ wt,
                          int k, int kpad) {
  int i = blockIdx.x * blockDim.x + threadIdx.x;
  if (i >= 128 * kpad) return;
  int nn = i / kpad, c = i % kpad;
  wt[i] = (c < k) ? f32_to_bf16_rne(w[(long long)c * 128 + nn]) : (unsigned short)0;
}

// ---------------- WMMA GEMM:  C[M,128] = A[M,Kpad] * Bt^T ----------------
// Block: 256 threads = 8 waves laid out as 4 row-groups x 2 col-groups.
// Each wave: 32 rows x 64 cols = 2x4 grid of 16x16 WMMA tiles.
// A frag (16-bit A 16x32): lane<16 -> K {k0..k0+7, k0+16..k0+23},
//                          lane>=16 -> +8.  Two b128 loads per frag.
// B frag (16-bit B 32x16): lane = column n, same K order as A:
//                          lanes 0-15 -> K {k0..k0+7, k0+16..k0+23}, lanes 16-31 -> +8.
__global__ void __launch_bounds__(256)
k_gemm_wmma_bf16(const unsigned short* __restrict__ A,
                 const unsigned short* __restrict__ Bt,
                 float* __restrict__ C, int M, int Kpad) {
  extern __shared__ unsigned char smem[];
  unsigned short* sB = (unsigned short*)smem;

  const int tid    = threadIdx.x;
  const int lane   = tid & 31;
  const int wave   = tid >> 5;
  const int l16    = lane & 15;
  const int hi     = (lane >> 4) & 1;
  const int rowGrp = wave >> 1;   // 0..3 -> which 32-row strip
  const int colGrp = wave & 1;    // 0..1 -> which 64-col strip

  { // cooperative stage of Bt (128 x Kpad bf16) into LDS
    const uint4* g4 = (const uint4*)Bt;
    uint4* s4 = (uint4*)smem;
    const int n4 = (128 * Kpad) >> 3;
    for (int i = tid; i < n4; i += 256) s4[i] = g4[i];
  }
  __syncthreads();

  const long long mBase = (long long)blockIdx.x * 128 + rowGrp * 32;
  const unsigned short* aBase[2];
#pragma unroll
  for (int mi = 0; mi < 2; ++mi) {
    long long row = mBase + mi * 16 + l16;
    if (row >= M) row = M - 1;  // clamp (stores are guarded)
    aBase[mi] = A + row * (long long)Kpad + (hi ? 8 : 0);
  }

  v8f acc[2][4] = {};
  for (int k0 = 0; k0 < Kpad; k0 += 32) {
    FragAB a[2];
#pragma unroll
    for (int mi = 0; mi < 2; ++mi) {
      a[mi].q[0] = *(const uint4*)(aBase[mi] + k0);
      a[mi].q[1] = *(const uint4*)(aBase[mi] + k0 + 16);
    }
#pragma unroll
    for (int nj = 0; nj < 4; ++nj) {
      const unsigned short* bp =
          sB + (colGrp * 64 + nj * 16 + l16) * Kpad + k0 + (hi ? 8 : 0);
      FragAB b;
      b.q[0] = *(const uint4*)(bp);
      b.q[1] = *(const uint4*)(bp + 16);
      acc[0][nj] = __builtin_amdgcn_wmma_f32_16x16x32_bf16(
          false, a[0].v, false, b.v, (short)0, acc[0][nj], false, false);
      acc[1][nj] = __builtin_amdgcn_wmma_f32_16x16x32_bf16(
          false, a[1].v, false, b.v, (short)0, acc[1][nj], false, false);
    }
  }

  // C/D layout: lanes 0-15 -> M = r, N = lane; lanes 16-31 -> M = 8+r.
  const int rbase = hi ? 8 : 0;
#pragma unroll
  for (int mi = 0; mi < 2; ++mi) {
#pragma unroll
    for (int nj = 0; nj < 4; ++nj) {
#pragma unroll
      for (int r = 0; r < 8; ++r) {
        long long row = mBase + mi * 16 + rbase + r;
        if (row < M) C[row * 128 + colGrp * 64 + nj * 16 + l16] = acc[mi][nj][r];
      }
    }
  }
}

// ---------------- aggregation ----------------
// agg[i][:] = h[i][:] * dinv[i]   (self-loop term; also clears buffer)
__global__ void k_self_init(const float* __restrict__ h, const float* __restrict__ dinv,
                            float* __restrict__ agg, int n) {
  long long i = (long long)blockIdx.x * blockDim.x + threadIdx.x;
  if (i < (long long)n * 128) agg[i] = h[i] * dinv[i >> 7];
}
// one wave per edge: 32 lanes x float4 covers the 128-feature row
__global__ void k_scatter(const float* __restrict__ h, const int* __restrict__ src,
                          const int* __restrict__ dst, const float* __restrict__ dsq,
                          float* __restrict__ agg, int e) {
  long long idx = (long long)blockIdx.x * blockDim.x + threadIdx.x;
  if (idx >= (long long)e * 32) return;
  int ei = (int)(idx >> 5), c = (int)(idx & 31);
  int s = src[ei], d = dst[ei];
  float norm = dsq[s] * dsq[d];
  float4 v = ((const float4*)(h + (long long)s * 128))[c];
  float* ap = agg + (long long)d * 128 + c * 4;
  atomicAdd(ap + 0, v.x * norm);
  atomicAdd(ap + 1, v.y * norm);
  atomicAdd(ap + 2, v.z * norm);
  atomicAdd(ap + 3, v.w * norm);
}
// relu(agg + b) -> bf16 A matrix for the next GEMM (Kpad == 128)
__global__ void k_fin_bf16(const float* __restrict__ agg, const float* __restrict__ b,
                           unsigned short* __restrict__ a2, int n) {
  long long i = (long long)blockIdx.x * blockDim.x + threadIdx.x;
  if (i >= (long long)n * 128) return;
  float v = agg[i] + b[i & 127];
  a2[i] = f32_to_bf16_rne(v > 0.0f ? v : 0.0f);
}
// relu(agg + b) in place (layer 2 output, f32 for pooling)
__global__ void k_fin_f32(float* __restrict__ agg, const float* __restrict__ b, int n) {
  long long i = (long long)blockIdx.x * blockDim.x + threadIdx.x;
  if (i >= (long long)n * 128) return;
  float v = agg[i] + b[i & 127];
  agg[i] = v > 0.0f ? v : 0.0f;
}

// ---------------- pooling + FC head ----------------
__global__ void k_pool_zero(float* sums, float* cnts) {
  int i = blockIdx.x * blockDim.x + threadIdx.x;
  if (i < 256 * 128) sums[i] = 0.0f;
  if (i < 256) cnts[i] = 0.0f;
}
__global__ void k_pool_add(const float* __restrict__ h, const int* __restrict__ batch,
                           float* __restrict__ sums, float* __restrict__ cnts, int n) {
  long long idx = (long long)blockIdx.x * blockDim.x + threadIdx.x;
  if (idx >= (long long)n * 32) return;
  int i = (int)(idx >> 5), c = (int)(idx & 31);
  int g = batch[i];
  float4 v = ((const float4*)(h + (long long)i * 128))[c];
  float* sp = sums + g * 128 + c * 4;
  atomicAdd(sp + 0, v.x);
  atomicAdd(sp + 1, v.y);
  atomicAdd(sp + 2, v.z);
  atomicAdd(sp + 3, v.w);
  if (c == 0) atomicAdd(&cnts[g], 1.0f);
}
// block per graph: mean -> relu(mean @ Wf1 + bf1)
__global__ void k_fc1(const float* __restrict__ sums, const float* __restrict__ cnts,
                      const float* __restrict__ Wf1, const float* __restrict__ bf1,
                      float* __restrict__ g1) {
  __shared__ float mean[128];
  int g = blockIdx.x, t = threadIdx.x;
  float inv = 1.0f / fmaxf(cnts[g], 1.0f);
  for (int k = t; k < 128; k += 64) mean[k] = sums[g * 128 + k] * inv;
  __syncthreads();
  float acc = bf1[t];
  for (int k = 0; k < 128; ++k) acc += mean[k] * Wf1[k * 64 + t];
  g1[g * 64 + t] = acc > 0.0f ? acc : 0.0f;
}
__global__ void k_fc2(const float* __restrict__ g1, const float* __restrict__ Wf2,
                      const float* __restrict__ bf2, float* __restrict__ out) {
  int idx = blockIdx.x * blockDim.x + threadIdx.x;
  if (idx >= 256 * 2) return;
  int g = idx >> 1, o = idx & 1;
  float acc = bf2[o];
  for (int j = 0; j < 64; ++j) acc += g1[g * 64 + j] * Wf2[j * 2 + o];
  out[idx] = acc;
}

// ---------------------------------------------------------------------------
extern "C" void kernel_launch(void* const* d_in, const int* in_sizes, int n_in,
                              void* d_out, int out_size, void* d_ws, size_t ws_size,
                              hipStream_t stream) {
  const float* x      = (const float*)d_in[0];
  const int*   eidx   = (const int*)d_in[1];
  const int*   batch  = (const int*)d_in[2];
  const float* W1     = (const float*)d_in[3];
  const float* b1     = (const float*)d_in[4];
  const float* W2     = (const float*)d_in[5];
  const float* b2     = (const float*)d_in[6];
  const float* Wf1    = (const float*)d_in[7];
  const float* bf1    = (const float*)d_in[8];
  const float* Wf2    = (const float*)d_in[9];
  const float* bf2    = (const float*)d_in[10];
  float* out = (float*)d_out;

  const int N = in_sizes[2];        // 100000 nodes
  const int E = in_sizes[1] / 2;    // 1.6M edges
  const int K1 = 205, KP1 = 224, KP2 = 128;
  const int* src = eidx;
  const int* dst = eidx + E;

  // workspace carve-up (256B aligned)
  size_t off = 0;
  auto carve = [&](size_t bytes) -> void* {
    void* p = (char*)d_ws + off;
    off += (bytes + 255) & ~(size_t)255;
    return p;
  };
  float*          dsq  = (float*)carve((size_t)N * 4);           // rsqrt(deg)
  float*          dinv = (float*)carve((size_t)N * 4);           // 1/deg
  unsigned short* Abuf = (unsigned short*)carve((size_t)N * KP1 * 2);
  unsigned short* Wt   = (unsigned short*)carve((size_t)128 * KP1 * 2);
  float*          hx   = (float*)carve((size_t)N * 128 * 4);     // GEMM out
  float*          agg  = (float*)carve((size_t)N * 128 * 4);     // aggregation
  float*          psum = (float*)carve((size_t)256 * 128 * 4);
  float*          pcnt = (float*)carve((size_t)256 * 4);
  float*          g1   = (float*)carve((size_t)256 * 64 * 4);
  (void)ws_size; (void)n_in; (void)out_size;

  const int B = 256;
  auto blocks = [](long long work, int b) { return (unsigned)((work + b - 1) / b); };
  const unsigned gM = blocks(N, 128);  // GEMM row-blocks

  // degree / normalization
  k_deg_init<<<blocks(N, B), B, 0, stream>>>(dsq, N);
  k_deg_count<<<blocks(E, B), B, 0, stream>>>(dst, dsq, E);
  k_deg_fin<<<blocks(N, B), B, 0, stream>>>(dsq, dinv, N);

  // ---- layer 1 ----
  k_conv_a<<<blocks((long long)N * KP1, B), B, 0, stream>>>(x, Abuf, N, K1, KP1);
  k_conv_wt<<<blocks(128 * KP1, B), B, 0, stream>>>(W1, Wt, K1, KP1);
  k_gemm_wmma_bf16<<<gM, B, 128 * KP1 * 2, stream>>>(Abuf, Wt, hx, N, KP1);
  k_self_init<<<blocks((long long)N * 128, B), B, 0, stream>>>(hx, dinv, agg, N);
  k_scatter<<<blocks((long long)E * 32, B), B, 0, stream>>>(hx, src, dst, dsq, agg, E);
  k_fin_bf16<<<blocks((long long)N * 128, B), B, 0, stream>>>(agg, b1, Abuf, N);

  // ---- layer 2 ----
  k_conv_wt<<<blocks(128 * KP2, B), B, 0, stream>>>(W2, Wt, KP2, KP2);
  k_gemm_wmma_bf16<<<gM, B, 128 * KP2 * 2, stream>>>(Abuf, Wt, hx, N, KP2);
  k_self_init<<<blocks((long long)N * 128, B), B, 0, stream>>>(hx, dinv, agg, N);
  k_scatter<<<blocks((long long)E * 32, B), B, 0, stream>>>(hx, src, dst, dsq, agg, E);
  k_fin_f32<<<blocks((long long)N * 128, B), B, 0, stream>>>(agg, b2, N);

  // ---- pool + FC head ----
  k_pool_zero<<<blocks(256 * 128, B), B, 0, stream>>>(psum, pcnt);
  k_pool_add<<<blocks((long long)N * 32, B), B, 0, stream>>>(agg, batch, psum, pcnt, N);
  k_fc1<<<256, 64, 0, stream>>>(psum, pcnt, Wf1, bf1, g1);
  k_fc2<<<2, 256, 0, stream>>>(g1, Wf2, bf2, out);
}
